// XFeat_36627481100772
// MI455X (gfx1250) — matrix operand order repaired and checked by reading, not compiled, not run on hardware
//
#include <hip/hip_runtime.h>
#include <hip/hip_bf16.h>
#include <math.h>

// ---------------- problem constants (from reference) ----------------
#define BB   4
#define NN   6144
#define DD   64
#define HIDD 512
#define OUTD 64
#define PTS  (BB * NN)            // 24576 points
#define FINE_CONF 0.25f

typedef __attribute__((ext_vector_type(16))) _Float16 v16h;
typedef __attribute__((ext_vector_type(8)))  float    v8f;

union AF {                 // one WMMA 16-bit operand fragment (8 VGPRs / lane)
    v16h v;
    _Float16 h[16];
    uint4 q[2];
};

union H4 {                 // 4 halves <-> one 64-bit chunk (for LDS staging)
    _Float16 h[4];
    unsigned long long u;
};

// order-preserving float -> u32 key (so u64 max == float max)
__device__ __forceinline__ unsigned int okey(float f) {
    unsigned int u = __float_as_uint(f);
    return (u & 0x80000000u) ? ~u : (u | 0x80000000u);
}

// =====================================================================
// 0) zero the packed-argmax arrays (ws is poisoned 0xAA by the harness)
// =====================================================================
__global__ void zero_u64_kernel(unsigned long long* p, int count) {
    int i = blockIdx.x * blockDim.x + threadIdx.x;
    if (i < count) p[i] = 0ull;
}

// =====================================================================
// 1) convert weights f32 [K x Nc] -> transposed f16 Wt [Nc x K]
// =====================================================================
__global__ void wt_convert_kernel(const float* __restrict__ src,
                                  _Float16* __restrict__ dst, int K, int Nc) {
    int i = blockIdx.x * blockDim.x + threadIdx.x;
    if (i >= K * Nc) return;
    int k = i / Nc, c = i % Nc;
    dst[(size_t)c * K + k] = (_Float16)src[i];
}

// =====================================================================
// 2) fused cosine-sim GEMM + bidirectional argmax (never materializes
//    the 604MB cossim matrix). One block = one 128x128 tile of one batch.
//    Two independent WMMA accumulators per subtile + pipelined B loads;
//    argmax winner selected with f32 compares, packed to u64 only once.
// =====================================================================
__global__ __launch_bounds__(256)
void sim_argmax_kernel(const float* __restrict__ f1, const float* __restrict__ f2,
                       unsigned long long* __restrict__ rowPk,
                       unsigned long long* __restrict__ colPk) {
    const int b = blockIdx.z, I = blockIdx.x, J = blockIdx.y;
    __shared__ _Float16 f1s[128 * 64];
    __shared__ _Float16 f2s[128 * 64];
    __shared__ unsigned long long colBest[128];

    const int tid = threadIdx.x;
    const size_t base1 = ((size_t)b * NN + (size_t)I * 128) * 64;
    const size_t base2 = ((size_t)b * NN + (size_t)J * 128) * 64;
    for (int i = tid * 4; i < 128 * 64; i += 256 * 4) {
        float4 v1 = *(const float4*)(f1 + base1 + i);
        float4 v2 = *(const float4*)(f2 + base2 + i);
        H4 t1, t2;
        t1.h[0] = (_Float16)v1.x; t1.h[1] = (_Float16)v1.y;
        t1.h[2] = (_Float16)v1.z; t1.h[3] = (_Float16)v1.w;
        t2.h[0] = (_Float16)v2.x; t2.h[1] = (_Float16)v2.y;
        t2.h[2] = (_Float16)v2.z; t2.h[3] = (_Float16)v2.w;
        *(unsigned long long*)&f1s[i] = t1.u;     // ds_store_b64
        *(unsigned long long*)&f2s[i] = t2.u;
    }
    if (tid < 128) colBest[tid] = 0ull;
    __syncthreads();

    const int wave = tid >> 5, lane = tid & 31;     // wave32
    const int m  = lane & 15;                       // M row / N col within tile
    const int hl = lane >> 4;                       // half-wave select
    const int kbA = hl * 8;                         // A-frag K sub-offset
    const int kbB = hl * 8;                         // B-frag K sub-offset

    // A fragments for this wave's 16-row strip, K=0..31 and K=32..63
    AF a0, a1;
    const _Float16* ar = &f1s[(wave * 16 + m) * 64];
    a0.q[0] = *(const uint4*)(ar + kbA);       a0.q[1] = *(const uint4*)(ar + kbA + 16);
    a1.q[0] = *(const uint4*)(ar + 32 + kbA);  a1.q[1] = *(const uint4*)(ar + 32 + kbA + 16);

    float rbv[8]; int rbc[8];
#pragma unroll
    for (int r = 0; r < 8; ++r) { rbv[r] = -3.0e38f; rbc[r] = 0; }

    auto loadB = [&](int ct, AF& x0, AF& x1) {
        const _Float16* br = &f2s[(ct * 16 + m) * 64];   // B[k][n] = f2[col n][k]
        x0.q[0] = *(const uint4*)(br + kbB);       x0.q[1] = *(const uint4*)(br + kbB + 16);
        x1.q[0] = *(const uint4*)(br + 32 + kbB);  x1.q[1] = *(const uint4*)(br + 32 + kbB + 16);
    };

    AF b0c, b1c, b0n, b1n;
    loadB(0, b0c, b1c);
#pragma unroll 2
    for (int ct = 0; ct < 8; ++ct) {
        if (ct + 1 < 8) loadB(ct + 1, b0n, b1n);     // pipeline next B frags

        v8f s0 = {0, 0, 0, 0, 0, 0, 0, 0};
        v8f s1 = {0, 0, 0, 0, 0, 0, 0, 0};
        s0 = __builtin_amdgcn_wmma_f32_16x16x32_f16(false, a0.v, false, b0c.v,
                                                    (short)0, s0, false, false);
        s1 = __builtin_amdgcn_wmma_f32_16x16x32_f16(false, a1.v, false, b1c.v,
                                                    (short)0, s1, false, false);

        // cheap f32 winner selection; pack to u64 only once per subtile
        float cbv = -3.0e38f; int cbr = 0;
#pragma unroll
        for (int r = 0; r < 8; ++r) {
            float v = s0[r] + s1[r];
            if (v > rbv[r]) { rbv[r] = v; rbc[r] = ct * 16 + m; }   // row argmax
            if (v > cbv)    { cbv = v; cbr = r; }                   // col candidate
        }
        unsigned int grow = (unsigned int)(I * 128 + wave * 16 + cbr + 8 * hl);
        unsigned long long p =
            ((unsigned long long)okey(cbv) << 32) | (unsigned int)(0xFFFFFFFFu - grow);
        unsigned long long o = __shfl_xor(p, 16);    // combine the two half-waves
        if (o > p) p = o;
        if (hl == 0) atomicMax(&colBest[ct * 16 + m], p);   // ds_max_u64

        b0c = b0n; b1c = b1n;
    }

    // finalize row argmax: reduce 16 column-lanes, then global u64 atomic max
#pragma unroll
    for (int r = 0; r < 8; ++r) {
        unsigned long long p = ((unsigned long long)okey(rbv[r]) << 32) |
                               (unsigned int)(0xFFFFFFFFu - (unsigned int)(J * 128 + rbc[r]));
        for (int s = 8; s >= 1; s >>= 1) {
            unsigned long long o = __shfl_xor(p, s, 16);
            if (o > p) p = o;
        }
        if (m == 0) {
            int grow = I * 128 + wave * 16 + r + 8 * hl;
            atomicMax(&rowPk[(size_t)b * NN + grow], p);
        }
    }
    __syncthreads();
    if (tid < 128)
        atomicMax(&colPk[(size_t)b * NN + (size_t)J * 128 + tid], colBest[tid]);
}

// =====================================================================
// 3) decode matches, mutual check, gather feats2/kps2, build f16 MLP input
// =====================================================================
__global__ void match_build_kernel(const float* __restrict__ f1, const float* __restrict__ f2,
                                   const float* __restrict__ kps2,
                                   const unsigned long long* __restrict__ rowPk,
                                   const unsigned long long* __restrict__ colPk,
                                   int* __restrict__ mutualA, float* __restrict__ mk1,
                                   _Float16* __restrict__ Xout) {
    int p = blockIdx.x * blockDim.x + threadIdx.x;
    if (p >= PTS) return;
    int b = p / NN, n = p % NN;

    unsigned long long rp = rowPk[p];
    int m12 = (int)(0xFFFFFFFFu - (unsigned int)(rp & 0xFFFFFFFFull));
    unsigned long long cp = colPk[(size_t)b * NN + m12];
    int back = (int)(0xFFFFFFFFu - (unsigned int)(cp & 0xFFFFFFFFull));
    mutualA[p] = (back == n) ? 1 : 0;

    mk1[(size_t)p * 2 + 0] = kps2[((size_t)b * NN + m12) * 2 + 0];
    mk1[(size_t)p * 2 + 1] = kps2[((size_t)b * NN + m12) * 2 + 1];

    const float* s1 = f1 + (size_t)p * DD;
    const float* s2 = f2 + ((size_t)b * NN + m12) * DD;
    _Float16* xo = Xout + (size_t)p * (2 * DD);
#pragma unroll
    for (int i = 0; i < DD; i += 4) {
        float4 v1 = *(const float4*)(s1 + i);
        float4 v2 = *(const float4*)(s2 + i);
        H4 t1, t2;
        t1.h[0] = (_Float16)v1.x; t1.h[1] = (_Float16)v1.y;
        t1.h[2] = (_Float16)v1.z; t1.h[3] = (_Float16)v1.w;
        t2.h[0] = (_Float16)v2.x; t2.h[1] = (_Float16)v2.y;
        t2.h[2] = (_Float16)v2.z; t2.h[3] = (_Float16)v2.w;
        *(unsigned long long*)&xo[i]      = t1.u;
        *(unsigned long long*)&xo[DD + i] = t2.u;
    }
}

// =====================================================================
// 4) WMMA GEMM: out = act(X[R x K] * W[K x Nc] + bias), W transposed f16
//    (Wt[Nc x K]). Block = 128 rows x (NTILES*16) cols, 8 waves, wave =
//    16-row strip. K/NC/NTILES compile-time so nt-loop fully unrolls and
//    all 8 B-fragments are in flight before the wait. L2-resident data.
// =====================================================================
template <int K, int NC, int NTILES, bool RELU, bool F32OUT>
__global__ __launch_bounds__(256)
void gemm_bias_act(const _Float16* __restrict__ X, const _Float16* __restrict__ Wt,
                   const float* __restrict__ bias, _Float16* __restrict__ outH,
                   float* __restrict__ outF) {
    const int rowBase = blockIdx.x * 128;
    const int nBase   = blockIdx.y * (NTILES * 16);
    const int tid  = threadIdx.x;
    const int wave = tid >> 5, lane = tid & 31;
    const int m  = lane & 15;
    const int hl = lane >> 4;
    const int kbA = hl * 8, kbB = hl * 8;

    v8f acc[NTILES];
#pragma unroll
    for (int i = 0; i < NTILES; ++i) acc[i] = (v8f){0, 0, 0, 0, 0, 0, 0, 0};

    const _Float16* xr = X + (size_t)(rowBase + wave * 16 + m) * K;
    const _Float16* wr = Wt + (size_t)(nBase + m) * K;
#pragma unroll 2
    for (int kt = 0; kt < (K >> 5); ++kt) {
        AF a;
        const _Float16* xp = xr + kt * 32;
        a.q[0] = *(const uint4*)(xp + kbA);
        a.q[1] = *(const uint4*)(xp + kbA + 16);
        __builtin_prefetch(xp + 32, 0, 1);                 // global_prefetch_b8

        AF bb[NTILES];
#pragma unroll
        for (int nt = 0; nt < NTILES; ++nt) {              // batch all B loads
            const _Float16* wp = wr + (size_t)nt * 16 * K + kt * 32;
            bb[nt].q[0] = *(const uint4*)(wp + kbB);
            bb[nt].q[1] = *(const uint4*)(wp + kbB + 16);
        }
#pragma unroll
        for (int nt = 0; nt < NTILES; ++nt)
            acc[nt] = __builtin_amdgcn_wmma_f32_16x16x32_f16(false, a.v, false, bb[nt].v,
                                                             (short)0, acc[nt], false, false);
    }

#pragma unroll
    for (int nt = 0; nt < NTILES; ++nt) {
        int col = nBase + nt * 16 + m;
        float bv = bias[col];
#pragma unroll
        for (int r = 0; r < 8; ++r) {
            int row = rowBase + wave * 16 + r + 8 * hl;
            float v = acc[nt][r] + bv;
            if (RELU && v < 0.f) v = 0.f;
            if (F32OUT) outF[(size_t)row * NC + col] = v;
            else        outH[(size_t)row * NC + col] = (_Float16)v;
        }
    }
}

// =====================================================================
// 5) softmax(3x) heat -> conf + subpixel offsets -> matches + mask
// =====================================================================
__global__ void final_kernel(const float* __restrict__ logits, const float* __restrict__ kps1,
                             const float* __restrict__ scales1, const float* __restrict__ mk1,
                             const int* __restrict__ mutualA, float* __restrict__ out) {
    int p = blockIdx.x * blockDim.x + threadIdx.x;
    if (p >= PTS) return;
    const float* L = logits + (size_t)p * OUTD;

    float mx = -3.0e38f;
#pragma unroll
    for (int i = 0; i < OUTD; ++i) mx = fmaxf(mx, L[i]);

    float sum = 0.f, cx = 0.f, cy = 0.f;
#pragma unroll
    for (int i = 0; i < OUTD; ++i) {
        float e = expf(3.0f * (L[i] - mx));
        sum += e;
        cx += e * (float)((i & 7) - 4);
        cy += e * (float)((i >> 3) - 4);
    }
    float inv  = 1.0f / sum;
    float conf = inv;                 // max(softmax) = exp(0)/sum
    float s = scales1[p];
    float m0x = kps1[(size_t)p * 2 + 0] + cx * inv * s;
    float m0y = kps1[(size_t)p * 2 + 1] + cy * inv * s;

    out[(size_t)p * 4 + 0] = m0x;
    out[(size_t)p * 4 + 1] = m0y;
    out[(size_t)p * 4 + 2] = mk1[(size_t)p * 2 + 0];
    out[(size_t)p * 4 + 3] = mk1[(size_t)p * 2 + 1];
    out[(size_t)PTS * 4 + p] = (mutualA[p] && conf > FINE_CONF) ? 1.0f : 0.0f;
}

// =====================================================================
// host-side orchestration
// =====================================================================
extern "C" void kernel_launch(void* const* d_in, const int* in_sizes, int n_in,
                              void* d_out, int out_size, void* d_ws, size_t ws_size,
                              hipStream_t stream) {
    const float* feats1  = (const float*)d_in[0];
    const float* feats2  = (const float*)d_in[1];
    const float* kps1    = (const float*)d_in[2];
    const float* kps2    = (const float*)d_in[3];
    const float* scales1 = (const float*)d_in[4];
    const float* w1 = (const float*)d_in[5];  const float* b1 = (const float*)d_in[6];
    const float* w2 = (const float*)d_in[7];  const float* b2 = (const float*)d_in[8];
    const float* w3 = (const float*)d_in[9];  const float* b3 = (const float*)d_in[10];
    const float* w4 = (const float*)d_in[11]; const float* b4 = (const float*)d_in[12];
    const float* w5 = (const float*)d_in[13]; const float* b5 = (const float*)d_in[14];

    char* ws = (char*)d_ws;
    size_t off = 0;
    unsigned long long* rowPk = (unsigned long long*)(ws + off); off += (size_t)PTS * 8;
    unsigned long long* colPk = (unsigned long long*)(ws + off); off += (size_t)PTS * 8;
    int*      mutualA = (int*)(ws + off);       off += (size_t)PTS * 4;
    float*    mk1     = (float*)(ws + off);     off += (size_t)PTS * 2 * 4;
    _Float16* X       = (_Float16*)(ws + off);  off += (size_t)PTS * 2 * DD * 2;
    _Float16* H1      = (_Float16*)(ws + off);  off += (size_t)PTS * HIDD * 2;
    _Float16* H2      = (_Float16*)(ws + off);  off += (size_t)PTS * HIDD * 2;
    float*    logits  = (float*)(ws + off);     off += (size_t)PTS * OUTD * 4;
    _Float16* wt1     = (_Float16*)(ws + off);  off += (size_t)HIDD * 2 * DD * 2;
    _Float16* wt2     = (_Float16*)(ws + off);  off += (size_t)HIDD * HIDD * 2;
    _Float16* wt3     = (_Float16*)(ws + off);  off += (size_t)HIDD * HIDD * 2;
    _Float16* wt4     = (_Float16*)(ws + off);  off += (size_t)HIDD * HIDD * 2;
    _Float16* wt5     = (_Float16*)(ws + off);  off += (size_t)OUTD * HIDD * 2;
    (void)ws_size; (void)n_in; (void)in_sizes; (void)out_size;

    // 0) zero packed-argmax arrays (row + col are contiguous)
    zero_u64_kernel<<<(2 * PTS + 255) / 256, 256, 0, stream>>>(rowPk, 2 * PTS);

    // 1) transpose+convert weights to f16
    wt_convert_kernel<<<(2 * DD * HIDD + 255) / 256, 256, 0, stream>>>(w1, wt1, 2 * DD, HIDD);
    wt_convert_kernel<<<(HIDD * HIDD + 255) / 256, 256, 0, stream>>>(w2, wt2, HIDD, HIDD);
    wt_convert_kernel<<<(HIDD * HIDD + 255) / 256, 256, 0, stream>>>(w3, wt3, HIDD, HIDD);
    wt_convert_kernel<<<(HIDD * HIDD + 255) / 256, 256, 0, stream>>>(w4, wt4, HIDD, HIDD);
    wt_convert_kernel<<<(HIDD * OUTD + 255) / 256, 256, 0, stream>>>(w5, wt5, HIDD, OUTD);

    // 2) fused cosine-sim + argmax (both directions)
    sim_argmax_kernel<<<dim3(NN / 128, NN / 128, BB), 256, 0, stream>>>(feats1, feats2,
                                                                        rowPk, colPk);

    // 3) mutual check + gathers + f16 MLP input
    match_build_kernel<<<(PTS + 255) / 256, 256, 0, stream>>>(feats1, feats2, kps2,
                                                              rowPk, colPk, mutualA, mk1, X);

    // 4) MLP: 128 -> 512 -> 512 -> 512 -> 512 -> 64
    gemm_bias_act<128, 512, 8, true, false>
        <<<dim3(PTS / 128, 4), 256, 0, stream>>>(X,  wt1, b1, H1, nullptr);
    gemm_bias_act<512, 512, 8, true, false>
        <<<dim3(PTS / 128, 4), 256, 0, stream>>>(H1, wt2, b2, H2, nullptr);
    gemm_bias_act<512, 512, 8, true, false>
        <<<dim3(PTS / 128, 4), 256, 0, stream>>>(H2, wt3, b3, H1, nullptr);
    gemm_bias_act<512, 512, 8, true, false>
        <<<dim3(PTS / 128, 4), 256, 0, stream>>>(H1, wt4, b4, H2, nullptr);
    gemm_bias_act<512, 64, 4, false, true>
        <<<dim3(PTS / 128, 1), 256, 0, stream>>>(H2, wt5, b5, nullptr, logits);

    // 5) softmax/subpixel/conf -> outputs
    final_kernel<<<(PTS + 255) / 256, 256, 0, stream>>>(logits, kps1, scales1, mk1, mutualA,
                                                        (float*)d_out);
}
